// StructuredTransformerDecoderBlock_10041633538563
// MI455X (gfx1250) — matrix operand order repaired and checked by reading, not compiled, not run on hardware
//
#include <hip/hip_runtime.h>
#include <hip/hip_bf16.h>

typedef __attribute__((ext_vector_type(16))) _Float16 v16h;
typedef __attribute__((ext_vector_type(8)))  _Float16 v8h;
typedef __attribute__((ext_vector_type(8)))  float    v8f;

// ---------------------------------------------------------------------------
// WMMA helpers (wave32; A and B share contiguous k = 16h + j, not the ISA table order)
// A 16x32 f16 tile from row-major f16 (ld in elements).
// Lane l: m = l&15, h = l>>4. Vector elem j: k = h*16 + j
__device__ __forceinline__ v16h load_a_tile(const _Float16* src, int ld, int lane) {
    int m = lane & 15, h = (lane >> 4) & 1;
    const _Float16* p = src + m * ld + h * 16;
    v8h lo = *(const v8h*)(p);
    v8h hi = *(const v8h*)(p + 8);
    v16h a;
#pragma unroll
    for (int j = 0; j < 8; ++j) { a[j] = lo[j]; a[8 + j] = hi[j]; }
    return a;
}

// B 32x16 f16 tile from TRANSPOSED weight layout [N][K] (ldT = K stride in elements).
__device__ __forceinline__ v16h load_bT_tile(const _Float16* srcT, int ldT, int lane) {
    int n = lane & 15, kb = (lane >> 4) << 4;
    return *(const v16h*)(srcT + n * ldT + kb);
}

// one 16x16 f32 tile, K = 32*ksteps, A row-major, B transposed [N][K]
__device__ __forceinline__ v8f gemm_tile(const _Float16* A, int lda,
                                         const _Float16* BT, int ldbT,
                                         int ksteps, int lane) {
    v8f c = {};
    for (int s = 0; s < ksteps; ++s) {
        v16h a = load_a_tile(A + s * 32, lda, lane);
        v16h b = load_bT_tile(BT + s * 32, ldbT, lane);
        c = __builtin_amdgcn_wmma_f32_16x16x32_f16(false, a, false, b,
                                                   (short)0, c, false, false);
    }
    return c;
}

// gfx1250 async global->LDS copy: 16 bytes per op; INST_OFFSET is added to BOTH
// the global address and the LDS address (ISA 10 / 15.18.3), tracked by ASYNCcnt.
#define ASYNC_CP16(ldsoff, gaddr, off)                                          \
    asm volatile("global_load_async_to_lds_b128 %0, %1, off offset:" #off       \
                 :: "v"(ldsoff), "v"(gaddr) : "memory")

__device__ __forceinline__ void wait_asynccnt0() {
    asm volatile("s_wait_asynccnt 0x0" ::: "memory");
}

// ---------------------------------------------------------------------------
// Kernel 1: Q projection + 3-layer MLP + msg = relu(mlp)+enc
// block = 256 threads (8 waves), 128 rows per block.
__global__ void k1_mlp(const float* __restrict__ features, const float* __restrict__ enc,
                       const float* __restrict__ W1, const float* __restrict__ b1,
                       const float* __restrict__ W2, const float* __restrict__ b2,
                       const float* __restrict__ W3, const float* __restrict__ b3,
                       const float* __restrict__ Wq, const float* __restrict__ bq,
                       float* __restrict__ q_ws, _Float16* __restrict__ msg_ws) {
    extern __shared__ char smem[];
    _Float16* sX = (_Float16*)smem;        // 128x128 activations (in)
    _Float16* sY = sX + 128 * 128;         // 128x128 activations (out)
    _Float16* sW = sY + 128 * 128;         // weight, transposed [n][k]
    float*    sB = (float*)(sW + 128 * 128);

    const int tid = threadIdx.x, lane = tid & 31, wave = tid >> 5;
    const int row0 = blockIdx.x * 128;

    // prefetch the encoder tile we'll need at the end (global_prefetch_b8)
    __builtin_prefetch(enc + (size_t)row0 * 128 + tid * 64, 0, 0);

    // stage raw features (f16) + Wq^T + bq
    for (int i = tid; i < 128 * 128; i += 256) sX[i] = (_Float16)features[row0 * 128 + i];
    for (int i = tid; i < 128 * 128; i += 256) {
        int k = i >> 7, n = i & 127;
        sW[n * 128 + k] = (_Float16)Wq[i];
    }
    if (tid < 128) sB[tid] = bq[tid];
    __syncthreads();

    // Q = X*Wq + bq   (raw X, no relu)
    {
        const _Float16* A = sX + wave * 16 * 128;
        for (int ct = 0; ct < 8; ++ct) {
            v8f c = gemm_tile(A, 128, sW + ct * 16 * 128, 128, 4, lane);
            int n = (lane & 15) + ct * 16, mb = ((lane >> 4) << 3) + wave * 16;
#pragma unroll
            for (int r = 0; r < 8; ++r)
                q_ws[(row0 + mb + r) * 128 + n] = c[r] + sB[n];
        }
    }
    __syncthreads();

    // relu in place
    for (int i = tid; i < 128 * 128; i += 256) {
        float f = (float)sX[i];
        sX[i] = (_Float16)(f > 0.f ? f : 0.f);
    }
    __syncthreads();

    // 3 MLP layers, all relu
    for (int L = 0; L < 3; ++L) {
        const float* Wl = (L == 0) ? W1 : (L == 1) ? W2 : W3;
        const float* bl = (L == 0) ? b1 : (L == 1) ? b2 : b3;
        for (int i = tid; i < 128 * 128; i += 256) {
            int k = i >> 7, n = i & 127;
            sW[n * 128 + k] = (_Float16)Wl[i];
        }
        if (tid < 128) sB[tid] = bl[tid];
        __syncthreads();
        const _Float16* A = sX + wave * 16 * 128;
        for (int ct = 0; ct < 8; ++ct) {
            v8f c = gemm_tile(A, 128, sW + ct * 16 * 128, 128, 4, lane);
            int n = (lane & 15) + ct * 16, mb = ((lane >> 4) << 3) + wave * 16;
#pragma unroll
            for (int r = 0; r < 8; ++r) {
                float v = c[r] + sB[n];
                sY[(mb + r) * 128 + n] = (_Float16)(v > 0.f ? v : 0.f);
            }
        }
        __syncthreads();
        _Float16* t = sX; sX = sY; sY = t;   // uniform per-thread pointer swap
    }

    // msg = local + encoder_features  (f16 workspace)
    for (int i = tid; i < 128 * 128; i += 256)
        msg_ws[row0 * 128 + i] = (_Float16)((float)sX[i] + enc[row0 * 128 + i]);
}

// ---------------------------------------------------------------------------
// Kernel 2: gather e-tiles (async to LDS), K/V projection via WMMA, attention.
// block = 256 threads, NT=4 nodes per block.
#define NT2 4
__global__ void k2_attn(const float* __restrict__ distances, const float* __restrict__ sequence,
                        const int* __restrict__ idx,
                        const float* __restrict__ Wk, const float* __restrict__ bk,
                        const float* __restrict__ Wv, const float* __restrict__ bv,
                        const float* __restrict__ q_ws, const _Float16* __restrict__ msg_ws,
                        _Float16* __restrict__ attnout_ws) {
    extern __shared__ char smem[];
    _Float16* sWk = (_Float16*)smem;            // transposed [128][192]
    _Float16* sWv = sWk + 128 * 192;            // transposed [128][192]
    _Float16* sE  = sWv + 128 * 192;            // [NT*32][192]
    _Float16* sK  = sE + NT2 * 32 * 192;        // [NT*32][128]
    _Float16* sV  = sK + NT2 * 32 * 128;        // [NT*32][128]
    float* sQ    = (float*)(sV + NT2 * 32 * 128);  // [NT][128]
    float* sAttn = sQ + NT2 * 128;                 // [NT*32][8]
    float* sBk   = sAttn + NT2 * 32 * 8;           // [128]
    float* sBv   = sBk + 128;                      // [128]

    const int tid = threadIdx.x, lane = tid & 31, wave = tid >> 5;
    const int node0 = blockIdx.x * NT2;

    // prefetch the distance/sequence slabs for this block (global_prefetch_b8)
    __builtin_prefetch(distances + (size_t)node0 * 32 * 32 + tid * 16, 0, 0);
    __builtin_prefetch(sequence + (size_t)node0 * 32 * 16 + tid * 8, 0, 0);

    // ---- async gather of neighbor msg rows into e-tiles ----
    // 128 rows x 256B ; 2 threads per row, 128B (8 x B128) each.
    {
        int row = tid >> 1, half = tid & 1;
        int nn = row >> 5, j = row & 31;
        int node = node0 + nn;
        int nbr = idx[node * 32 + j];
        const _Float16* gsrc = msg_ws + (size_t)nbr * 128 + half * 64;   // 128B chunk
        unsigned ldsoff = (unsigned)(size_t)(sE + row * 192 + half * 64); // flat->LDS low bits
        unsigned long long ga = (unsigned long long)(size_t)gsrc;
        ASYNC_CP16(ldsoff, ga, 0);
        ASYNC_CP16(ldsoff, ga, 16);
        ASYNC_CP16(ldsoff, ga, 32);
        ASYNC_CP16(ldsoff, ga, 48);
        ASYNC_CP16(ldsoff, ga, 64);
        ASYNC_CP16(ldsoff, ga, 80);
        ASYNC_CP16(ldsoff, ga, 96);
        ASYNC_CP16(ldsoff, ga, 112);
    }

    // stage weights transposed, K padded 176 -> 192 with zeros (overlaps with async)
    for (int i = tid; i < 176 * 128; i += 256) {
        int k = i >> 7, n = i & 127;
        sWk[n * 192 + k] = (_Float16)Wk[i];
        sWv[n * 192 + k] = (_Float16)Wv[i];
    }
    for (int i = tid; i < 16 * 128; i += 256) {
        int k = 176 + (i >> 7), n = i & 127;
        sWk[n * 192 + k] = (_Float16)0.f;
        sWv[n * 192 + k] = (_Float16)0.f;
    }
    if (tid < 128) { sBk[tid] = bk[tid]; sBv[tid] = bv[tid]; }
    for (int i = tid; i < NT2 * 128; i += 256) sQ[i] = q_ws[node0 * 128 + i];

    // dist/seq/pad columns of e (need f32->f16 conversion, VALU path)
    {
        int row = tid >> 1, half = tid & 1;
        int nn = row >> 5, j = row & 31;
        int node = node0 + nn;
        _Float16* er = sE + row * 192;
        if (half) {
            const float* d = distances + (node * 32 + j) * 32;
            for (int c = 0; c < 32; ++c) er[128 + c] = (_Float16)d[c];
            const float* s = sequence + (node * 32 + j) * 16;
            for (int c = 0; c < 16; ++c) er[160 + c] = (_Float16)s[c];
            for (int c = 176; c < 192; ++c) er[c] = (_Float16)0.f;
        }
    }
    wait_asynccnt0();     // every wave drains its async LDS writes
    __syncthreads();

    // K and V projections: 128 rows x 128 cols each -> 128 tiles, 16/wave
    for (int t = wave; t < 128; t += 8) {
        int mat = t >> 6, tile = t & 63, rt = tile >> 3, ct = tile & 7;
        const _Float16* A  = sE + rt * 16 * 192;
        const _Float16* BT = (mat ? sWv : sWk) + ct * 16 * 192;
        v8f c = gemm_tile(A, 192, BT, 192, 6, lane);
        const float* bias = mat ? sBv : sBk;
        _Float16* dst = mat ? sV : sK;
        int n = (lane & 15) + ct * 16, mb = ((lane >> 4) << 3) + rt * 16;
#pragma unroll
        for (int r = 0; r < 8; ++r)
            dst[(mb + r) * 128 + n] = (_Float16)(c[r] + bias[n]);
    }
    __syncthreads();

    // attention: 2 waves per node, 4 heads per wave; lane = neighbor (wave32)
    const int nn = wave >> 1;
    const int h0 = (wave & 1) * 4;
    const int j = lane;
    float lg[4];
#pragma unroll
    for (int hh = 0; hh < 4; ++hh) {
        int h = h0 + hh;
        const float* qh = sQ + nn * 128 + h * 16;
        const _Float16* kr = sK + (nn * 32 + j) * 128 + h * 16;
        float dot = 0.f;
#pragma unroll
        for (int a = 0; a < 16; ++a) dot += qh[a] * (float)kr[a];
        lg[hh] = dot * 0.25f;                       // 1/sqrt(ADIM=16)
    }
#pragma unroll
    for (int hh = 0; hh < 4; ++hh) {
        float m = lg[hh];
        for (int off = 16; off; off >>= 1) m = fmaxf(m, __shfl_xor(m, off, 32));
        float e = __expf(lg[hh] - m);
        float s = e;
        for (int off = 16; off; off >>= 1) s += __shfl_xor(s, off, 32);
        sAttn[(nn * 32 + j) * 8 + h0 + hh] = e / s;
    }
    __syncthreads();

    // out = sum_k attn * v ; 64 outputs per wave, 2 per lane
    for (int o = lane; o < 64; o += 32) {
        int h = h0 + (o >> 4), a = o & 15;
        float acc = 0.f;
        for (int jj = 0; jj < 32; ++jj)
            acc += sAttn[(nn * 32 + jj) * 8 + h] * (float)sV[(nn * 32 + jj) * 128 + h * 16 + a];
        attnout_ws[(size_t)(node0 + nn) * 128 + h * 16 + a] = (_Float16)acc;
    }
}

// ---------------------------------------------------------------------------
// Kernel 3: d_out = features + out*Wo + bo
__global__ void k3_out(const float* __restrict__ features,
                       const float* __restrict__ Wo, const float* __restrict__ bo,
                       const _Float16* __restrict__ attnout_ws, float* __restrict__ out) {
    extern __shared__ char smem[];
    _Float16* sA = (_Float16*)smem;        // 128x128
    _Float16* sW = sA + 128 * 128;         // Wo transposed
    float*    sB = (float*)(sW + 128 * 128);
    const int tid = threadIdx.x, lane = tid & 31, wave = tid >> 5;
    const int row0 = blockIdx.x * 128;

    __builtin_prefetch(features + (size_t)row0 * 128 + tid * 64, 0, 0);

    for (int i = tid; i < 128 * 128; i += 256) sA[i] = attnout_ws[(size_t)row0 * 128 + i];
    for (int i = tid; i < 128 * 128; i += 256) {
        int k = i >> 7, n = i & 127;
        sW[n * 128 + k] = (_Float16)Wo[i];
    }
    if (tid < 128) sB[tid] = bo[tid];
    __syncthreads();

    const _Float16* A = sA + wave * 16 * 128;
    for (int ct = 0; ct < 8; ++ct) {
        v8f c = gemm_tile(A, 128, sW + ct * 16 * 128, 128, 4, lane);
        int n = (lane & 15) + ct * 16, mb = ((lane >> 4) << 3) + wave * 16;
#pragma unroll
        for (int r = 0; r < 8; ++r) {
            int gr = row0 + mb + r;
            out[gr * 128 + n] = features[gr * 128 + n] + c[r] + sB[n];
        }
    }
}

// ---------------------------------------------------------------------------
extern "C" void kernel_launch(void* const* d_in, const int* in_sizes, int n_in,
                              void* d_out, int out_size, void* d_ws, size_t ws_size,
                              hipStream_t stream) {
    const float* features = (const float*)d_in[0];
    const float* distances = (const float*)d_in[1];
    const float* sequence = (const float*)d_in[2];
    const float* enc      = (const float*)d_in[3];
    const int*   idx      = (const int*)d_in[4];
    const float* W1 = (const float*)d_in[5];  const float* b1 = (const float*)d_in[6];
    const float* W2 = (const float*)d_in[7];  const float* b2 = (const float*)d_in[8];
    const float* W3 = (const float*)d_in[9];  const float* b3 = (const float*)d_in[10];
    const float* Wq = (const float*)d_in[11]; const float* bq = (const float*)d_in[12];
    const float* Wk = (const float*)d_in[13]; const float* bk = (const float*)d_in[14];
    const float* Wv = (const float*)d_in[15]; const float* bv = (const float*)d_in[16];
    const float* Wo = (const float*)d_in[17]; const float* bo = (const float*)d_in[18];
    float* out = (float*)d_out;

    const int N = in_sizes[0] / 128;   // 16384

    // workspace layout
    float*     q_ws       = (float*)d_ws;                                  // N*128 f32
    _Float16*  msg_ws     = (_Float16*)((char*)d_ws + (size_t)N * 128 * 4);// N*128 f16
    _Float16*  attnout_ws = msg_ws + (size_t)N * 128;                      // N*128 f16

    const size_t shm1 = (size_t)3 * 128 * 128 * 2 + 128 * 4;               // ~98.8 KB
    const size_t shm2 = (size_t)2 * 128 * 192 * 2 + (size_t)NT2 * 32 * 192 * 2
                      + (size_t)2 * NT2 * 32 * 128 * 2
                      + (size_t)NT2 * 128 * 4 + (size_t)NT2 * 32 * 8 * 4 + 2 * 128 * 4; // ~215 KB
    const size_t shm3 = (size_t)2 * 128 * 128 * 2 + 128 * 4;               // ~64.5 KB

    (void)hipFuncSetAttribute((const void*)k1_mlp, hipFuncAttributeMaxDynamicSharedMemorySize, (int)shm1);
    (void)hipFuncSetAttribute((const void*)k2_attn, hipFuncAttributeMaxDynamicSharedMemorySize, (int)shm2);
    (void)hipFuncSetAttribute((const void*)k3_out, hipFuncAttributeMaxDynamicSharedMemorySize, (int)shm3);

    k1_mlp<<<N / 128, 256, shm1, stream>>>(features, enc, W1, b1, W2, b2, W3, b3,
                                           Wq, bq, q_ws, msg_ws);
    k2_attn<<<N / NT2, 256, shm2, stream>>>(distances, sequence, idx, Wk, bk, Wv, bv,
                                            q_ws, msg_ws, attnout_ws);
    k3_out<<<N / 128, 256, shm3, stream>>>(features, Wo, bo, attnout_ws, out);
}
